// NKF_1194000908672
// MI455X (gfx1250) — compile-verified, matrix operand order checked by reading, not run on hardware
//
#include <hip/hip_runtime.h>
#include <hip/hip_bf16.h>

#define B_ 32
#define T_ 512
#define N_ 256
#define K_ 64
#define H_ 64
#define NFLOW_ 4
#define HALF_ 128
#define TOK_ (B_ * T_)          // 16384
#define G4H_ 256                // 4*H
#define NP_ 1024                // N*(2+D)
#define LOG2PI_ 1.8378770664093453f

typedef __attribute__((ext_vector_type(16))) _Float16 v16h;
typedef __attribute__((ext_vector_type(8)))  float    v8f;

union H16 { v16h v; _Float16 h[16]; };
union F8  { v8f  v; float    f[8];  };

// A/B fragment element -> K index (16-bit 16x32 layout, ISA 7.12.2)
__device__ __forceinline__ int frag_k(int lane, int e) {
  return ((e >> 3) << 4) | (((lane >> 4) & 1) << 3) | (e & 7);
}

__device__ __forceinline__ v8f wmma16(v16h a, v16h b, v8f c) {
  return __builtin_amdgcn_wmma_f32_16x16x32_f16(false, a, false, b, (short)0, c,
                                                false, false);
}

__device__ __forceinline__ float sigm(float x) { return 1.0f / (1.0f + __expf(-x)); }

// Packed-fragment pointer: matrix with C cols, fragment (nt, kk), this lane.
__device__ __forceinline__ const _Float16*
pk(const _Float16* base, int KT, int nt, int kk, int lane) {
  return base + ((size_t)((nt * KT + kk) * 32 + lane) << 4);
}

// ---------------------------------------------------------------------------
// Kernel 0: pack a row-major f32 weight matrix (rows x cols, rows%16==0,
// cols%32==0) into WMMA B-fragment-major f16: [nt][kk][lane][16].
// Multi-layer matrices pack contiguously (layer stride preserved).
// ---------------------------------------------------------------------------
__global__ __launch_bounds__(256)
void pack_w_kernel(const float* __restrict__ W, int rows, int cols,
                   _Float16* __restrict__ dst) {
  const int KT = cols >> 5;
  const int total = rows * cols;
  for (int idx = blockIdx.x * blockDim.x + threadIdx.x; idx < total;
       idx += gridDim.x * blockDim.x) {
    int e = idx & 15;
    int lane = (idx >> 4) & 31;
    int blk = idx >> 9;              // nt*KT + kk
    int nt = blk / KT;
    int kk = blk - nt * KT;
    int n = nt * 16 + (lane & 15);
    int k = kk * 32 + frag_k(lane, e);
    dst[idx] = (_Float16)W[(size_t)n * cols + k];
  }
}

// ---------------------------------------------------------------------------
// Kernel 1: fused 4-layer inverse flow. One wave per 16-token tile, tile in LDS.
// ---------------------------------------------------------------------------
__global__ __launch_bounds__(64, 1)
void flow_inverse_kernel(const float* __restrict__ y,
                         const _Float16* __restrict__ pw1,
                         const _Float16* __restrict__ pw2,
                         const _Float16* __restrict__ pw3,
                         const float* __restrict__ b1,
                         const float* __restrict__ b2,
                         const float* __restrict__ b3,
                         float* __restrict__ z, float* __restrict__ logdet) {
  __shared__ float sy[2][16][N_];   // per-wave y tile
  __shared__ float sh[2][16][H_];   // per-wave hidden scratch
  __shared__ float sld[2][16];      // per-token log-det
  const int wave = threadIdx.x >> 5;
  const int lane = threadIdx.x & 31;
  const int tok0 = (blockIdx.x * 2 + wave) * 16;
  const int m = lane & 15;
  const int hi = lane >> 4;

  for (int i = lane; i < 16 * N_; i += 32)
    sy[wave][i >> 8][i & 255] = y[(size_t)tok0 * N_ + i];
  if (lane < 16) sld[wave][lane] = 0.0f;
  __syncthreads();

  bool flip = false;
  for (int j = 0; j < NFLOW_; ++j) {
    const int layer = NFLOW_ - 1 - j;
    if (j > 0) flip = !flip;
    const _Float16* W1 = pw1 + layer * H_ * HALF_;
    const _Float16* W2 = pw2 + layer * H_ * H_;
    const _Float16* W3 = pw3 + layer * N_ * H_;
    const float* B1 = b1 + layer * H_;
    const float* B2 = b2 + layer * H_;
    const float* B3 = b3 + layer * N_;

    F8 acc[16];

    // GEMM1: y1(16x128) @ w1^T(128x64)   KT=4, NT=4
#pragma unroll
    for (int nt = 0; nt < 4; ++nt)
#pragma unroll
      for (int q = 0; q < 8; ++q) acc[nt].f[q] = 0.0f;
#pragma unroll
    for (int kk = 0; kk < 4; ++kk) {
      H16 a;
#pragma unroll
      for (int e = 0; e < 16; ++e) {
        int kv = kk * 32 + frag_k(lane, e);
        int kp = flip ? (N_ - 1 - kv) : kv;
        a.h[e] = (_Float16)sy[wave][m][kp];
      }
#pragma unroll
      for (int nt = 0; nt < 4; ++nt) {
        H16 bf;
        bf.v = *(const v16h*)pk(W1, 4, nt, kk, lane);
        acc[nt].v = wmma16(a.v, bf.v, acc[nt].v);
      }
    }
    __syncthreads();
#pragma unroll
    for (int nt = 0; nt < 4; ++nt)
#pragma unroll
      for (int q = 0; q < 8; ++q) {
        int row = q + 8 * hi, n = nt * 16 + m;
        float vv = acc[nt].f[q] + B1[n];
        sh[wave][row][n] = vv > 0.0f ? vv : 0.0f;
      }
    __syncthreads();

    // GEMM2: h(16x64) @ w2^T(64x64)   KT=2, NT=4
#pragma unroll
    for (int nt = 0; nt < 4; ++nt)
#pragma unroll
      for (int q = 0; q < 8; ++q) acc[nt].f[q] = 0.0f;
#pragma unroll
    for (int kk = 0; kk < 2; ++kk) {
      H16 a;
#pragma unroll
      for (int e = 0; e < 16; ++e)
        a.h[e] = (_Float16)sh[wave][m][kk * 32 + frag_k(lane, e)];
#pragma unroll
      for (int nt = 0; nt < 4; ++nt) {
        H16 bf;
        bf.v = *(const v16h*)pk(W2, 2, nt, kk, lane);
        acc[nt].v = wmma16(a.v, bf.v, acc[nt].v);
      }
    }
    __syncthreads();
#pragma unroll
    for (int nt = 0; nt < 4; ++nt)
#pragma unroll
      for (int q = 0; q < 8; ++q) {
        int row = q + 8 * hi, n = nt * 16 + m;
        float vv = acc[nt].f[q] + B2[n];
        sh[wave][row][n] = vv > 0.0f ? vv : 0.0f;
      }
    __syncthreads();

    // GEMM3: h(16x64) @ w3^T(64x256)   KT=2, NT=16  -> s | t
#pragma unroll
    for (int nt = 0; nt < 16; ++nt)
#pragma unroll
      for (int q = 0; q < 8; ++q) acc[nt].f[q] = 0.0f;
#pragma unroll
    for (int kk = 0; kk < 2; ++kk) {
      H16 a;
#pragma unroll
      for (int e = 0; e < 16; ++e)
        a.h[e] = (_Float16)sh[wave][m][kk * 32 + frag_k(lane, e)];
#pragma unroll
      for (int nt = 0; nt < 16; ++nt) {
        H16 bf;
        bf.v = *(const v16h*)pk(W3, 2, nt, kk, lane);
        acc[nt].v = wmma16(a.v, bf.v, acc[nt].v);
      }
    }
    float rs[8] = {0.f, 0.f, 0.f, 0.f, 0.f, 0.f, 0.f, 0.f};
#pragma unroll
    for (int ct = 0; ct < 8; ++ct)
#pragma unroll
      for (int q = 0; q < 8; ++q) {
        int row = q + 8 * hi, c = ct * 16 + m;
        float s = tanhf(acc[ct].f[q] + B3[c]);
        float t = acc[8 + ct].f[q] + B3[HALF_ + c];
        int cv = HALF_ + c;
        int cp = flip ? (N_ - 1 - cv) : cv;
        float y2 = sy[wave][row][cp];
        sy[wave][row][cp] = (y2 - t) * __expf(-s);
        rs[q] -= s;
      }
#pragma unroll
    for (int q = 0; q < 8; ++q) atomicAdd(&sld[wave][q + 8 * hi], rs[q]);
    __syncthreads();
  }

  // final orientation: 3 flips -> flipped view is the result
  for (int i = lane; i < 16 * N_; i += 32) {
    int row = i >> 8, c = i & 255;
    int cp = flip ? (N_ - 1 - c) : c;
    z[(size_t)tok0 * N_ + i] = sy[wave][row][cp];
  }
  if (lane < 16) logdet[tok0 + lane] = sld[wave][lane];
}

// ---------------------------------------------------------------------------
// Kernel 2: input-gate projection inp @ W_ih^T + b_ih + b_hh, all timesteps.
// Output layout (T, B, 256). One wave per 16-token tile (tokens = t*B + b).
// A tile staged in LDS via coalesced float4 loads; W from packed f16.
// ---------------------------------------------------------------------------
__global__ __launch_bounds__(256, 1)
void gates_kernel(const float* __restrict__ inp, int inp_is_BT,
                  const _Float16* __restrict__ pWih,
                  const float* __restrict__ bih, const float* __restrict__ bhh,
                  float* __restrict__ out) {
  __shared__ float ta[8][16][K_];
  const int wave = threadIdx.x >> 5;
  const int lane = threadIdx.x & 31;
  const int tok0 = (blockIdx.x * 8 + wave) * 16;
  const int m = lane & 15;
  const int hi = lane >> 4;

  // stage A tile: lane handles half a row
  {
    int tok = tok0 + m;
    const float* rp = inp_is_BT
        ? (inp + ((size_t)(tok & (B_ - 1)) * T_ + (tok >> 5)) * K_)
        : (inp + (size_t)tok * K_);
#pragma unroll
    for (int c = 0; c < 32; c += 4) {
      float4 v = *(const float4*)(rp + hi * 32 + c);
      *(float4*)&ta[wave][m][hi * 32 + c] = v;
    }
  }
  __syncthreads();

  F8 acc[16];
#pragma unroll
  for (int nt = 0; nt < 16; ++nt)
#pragma unroll
    for (int q = 0; q < 8; ++q) acc[nt].f[q] = 0.0f;

#pragma unroll
  for (int kk = 0; kk < 2; ++kk) {
    H16 a;
#pragma unroll
    for (int e = 0; e < 16; ++e)
      a.h[e] = (_Float16)ta[wave][m][kk * 32 + frag_k(lane, e)];
#pragma unroll
    for (int nt = 0; nt < 16; ++nt) {
      H16 bf;
      bf.v = *(const v16h*)pk(pWih, 2, nt, kk, lane);
      acc[nt].v = wmma16(a.v, bf.v, acc[nt].v);
    }
  }
#pragma unroll
  for (int nt = 0; nt < 16; ++nt)
#pragma unroll
    for (int q = 0; q < 8; ++q) {
      int row = q + 8 * hi, n = nt * 16 + m;
      out[(size_t)(tok0 + row) * G4H_ + n] = acc[nt].f[q] + bih[n] + bhh[n];
    }
}

// ---------------------------------------------------------------------------
// Kernel 3: LSTM recurrence. One persistent 8-wave block; packed W_hh
// fragments in registers, h/c/gates in LDS; 8 WMMAs per wave per step.
// ---------------------------------------------------------------------------
__global__ __launch_bounds__(256, 1)
void lstm_rec_kernel(const float* __restrict__ xg,
                     const _Float16* __restrict__ pWhh,
                     float* __restrict__ hout) {
  __shared__ float sh_h[B_][H_];
  __shared__ float sc[B_][H_];
  __shared__ float sg[B_][G4H_];
  const int tid = threadIdx.x;
  const int wave = tid >> 5;
  const int lane = tid & 31;
  const int m = lane & 15;
  const int hi = lane >> 4;

  for (int i = tid; i < B_ * H_; i += 256) {
    sh_h[i >> 6][i & 63] = 0.0f;
    sc[i >> 6][i & 63] = 0.0f;
  }
  __syncthreads();

  // register-resident packed W_hh fragments: wave owns gate cols [32w, 32w+32)
  H16 bf[2][2];
#pragma unroll
  for (int nt = 0; nt < 2; ++nt)
#pragma unroll
    for (int kk = 0; kk < 2; ++kk)
      bf[nt][kk].v = *(const v16h*)pk(pWhh, 2, 2 * wave + nt, kk, lane);

  for (int t = 0; t < T_; ++t) {
    F8 acc[2][2];
#pragma unroll
    for (int mt = 0; mt < 2; ++mt)
#pragma unroll
      for (int nt = 0; nt < 2; ++nt)
#pragma unroll
        for (int q = 0; q < 8; ++q) acc[mt][nt].f[q] = 0.0f;
#pragma unroll
    for (int mt = 0; mt < 2; ++mt)
#pragma unroll
      for (int kk = 0; kk < 2; ++kk) {
        H16 a;
        int row = mt * 16 + m;
#pragma unroll
        for (int e = 0; e < 16; ++e)
          a.h[e] = (_Float16)sh_h[row][kk * 32 + frag_k(lane, e)];
#pragma unroll
        for (int nt = 0; nt < 2; ++nt)
          acc[mt][nt].v = wmma16(a.v, bf[nt][kk].v, acc[mt][nt].v);
      }
    const float* xgt = xg + (size_t)t * B_ * G4H_;
#pragma unroll
    for (int mt = 0; mt < 2; ++mt)
#pragma unroll
      for (int nt = 0; nt < 2; ++nt)
#pragma unroll
        for (int q = 0; q < 8; ++q) {
          int row = mt * 16 + q + 8 * hi;
          int n = (2 * wave + nt) * 16 + m;
          sg[row][n] = acc[mt][nt].f[q] + xgt[row * G4H_ + n];
        }
    __syncthreads();
#pragma unroll
    for (int e0 = 0; e0 < 8; ++e0) {
      int ei = tid * 8 + e0;
      int row = ei >> 6, hc = ei & 63;
      float gi = sg[row][hc];
      float gf = sg[row][H_ + hc];
      float gg = sg[row][2 * H_ + hc];
      float go = sg[row][3 * H_ + hc];
      float c = sigm(gf) * sc[row][hc] + sigm(gi) * tanhf(gg);
      float h = sigm(go) * tanhf(c);
      sc[row][hc] = c;
      sh_h[row][hc] = h;
      hout[((size_t)t * B_ + row) * H_ + hc] = h;
    }
    __syncthreads();
  }
}

// ---------------------------------------------------------------------------
// Kernel 4: head GEMM h2 @ head_w^T + head_b, exp() fused for var channels.
// ---------------------------------------------------------------------------
__global__ __launch_bounds__(256, 1)
void head_kernel(const float* __restrict__ h2,
                 const _Float16* __restrict__ pHw,
                 const float* __restrict__ hb, float* __restrict__ params) {
  __shared__ float ta[8][16][H_];
  const int wave = threadIdx.x >> 5;
  const int lane = threadIdx.x & 31;
  const int job = blockIdx.x * 8 + wave;  // 4096 jobs: 1024 tiles x 4 col slabs
  const int tile = job >> 2;
  const int slab = job & 3;
  const int tok0 = tile * 16;
  const int m = lane & 15;
  const int hi = lane >> 4;

  {
    const float* rp = h2 + (size_t)(tok0 + m) * H_;
#pragma unroll
    for (int c = 0; c < 32; c += 4) {
      float4 v = *(const float4*)(rp + hi * 32 + c);
      *(float4*)&ta[wave][m][hi * 32 + c] = v;
    }
  }
  __syncthreads();

  F8 acc[16];
#pragma unroll
  for (int nt = 0; nt < 16; ++nt)
#pragma unroll
    for (int q = 0; q < 8; ++q) acc[nt].f[q] = 0.0f;
#pragma unroll
  for (int kk = 0; kk < 2; ++kk) {
    H16 a;
#pragma unroll
    for (int e = 0; e < 16; ++e)
      a.h[e] = (_Float16)ta[wave][m][kk * 32 + frag_k(lane, e)];
#pragma unroll
    for (int nt = 0; nt < 16; ++nt) {
      H16 bf;
      bf.v = *(const v16h*)pk(pHw, 2, slab * 16 + nt, kk, lane);
      acc[nt].v = wmma16(a.v, bf.v, acc[nt].v);
    }
  }
#pragma unroll
  for (int nt = 0; nt < 16; ++nt)
#pragma unroll
    for (int q = 0; q < 8; ++q) {
      int row = q + 8 * hi;
      int n = slab * 256 + nt * 16 + m;
      float vv = acc[nt].f[q] + hb[n];
      if ((n & 3) < 2) vv = __expf(vv);  // obs_var / proc_var pre-exponentiated
      params[(size_t)(tok0 + row) * NP_ + n] = vv;
    }
}

// ---------------------------------------------------------------------------
// Kernel 5: Kalman scan. Block = batch b, thread = series n; 2x2 closed form.
// ---------------------------------------------------------------------------
__global__ __launch_bounds__(256, 1)
void kalman_kernel(const float* __restrict__ z, const float* __restrict__ params,
                   const float* __restrict__ logdet, float* __restrict__ perb) {
  __shared__ float red[256];
  const int b = blockIdx.x;
  const int n = threadIdx.x;
  float mu0 = 0.f, mu1 = 0.f;
  float p00 = 1.f, p01 = 0.f, p11 = 1.f;
  float ll = 0.f;
  const float* zb = z + (size_t)b * T_ * N_ + n;

  for (int t = 0; t < T_; ++t) {
    const float* pr = params + ((size_t)t * B_ + b) * NP_ + 4 * n;
    float4 pv4 = *(const float4*)pr;
    float ov = pv4.x, pv = pv4.y, a0 = pv4.z, a1 = pv4.w;
    float zt = zb[(size_t)t * N_];
    if (t + 1 < T_) {
      __builtin_prefetch(zb + (size_t)(t + 1) * N_, 0, 1);
      __builtin_prefetch(params + ((size_t)(t + 1) * B_ + b) * NP_ + 4 * n, 0, 1);
    }
    // predict (F = [[1,1],[0,1]])
    float mp0 = mu0 + mu1, mp1 = mu1;
    float q00 = p00 + 2.f * p01 + p11 + pv;
    float q01 = p01 + p11;
    float q11 = p11 + pv;
    // update
    float S = a0 * a0 * q00 + 2.f * a0 * a1 * q01 + a1 * a1 * q11 + ov;
    float rS = 1.f / S;
    float innov = zt - (a0 * mp0 + a1 * mp1);
    float k0 = (q00 * a0 + q01 * a1) * rS;
    float k1 = (q01 * a0 + q11 * a1) * rS;
    mu0 = mp0 + k0 * innov;
    mu1 = mp1 + k1 * innov;
    float atp0 = a0 * q00 + a1 * q01;
    float atp1 = a0 * q01 + a1 * q11;
    p00 = q00 - k0 * atp0;
    p01 = q01 - k0 * atp1;
    p11 = q11 - k1 * atp1;
    ll += -0.5f * (innov * innov * rS + __logf(S) + LOG2PI_);
  }
  // add this b's flow log-det (512 values over 256 threads) and reduce
  float ld = logdet[(size_t)b * T_ + n] + logdet[(size_t)b * T_ + 256 + n];
  red[n] = ll + ld;
  __syncthreads();
  for (int s = 128; s > 0; s >>= 1) {
    if (n < s) red[n] += red[n + s];
    __syncthreads();
  }
  if (n == 0) perb[b] = red[0];
}

__global__ void final_reduce_kernel(const float* __restrict__ perb,
                                    float* __restrict__ out) {
  __shared__ float red[32];
  int i = threadIdx.x;
  red[i] = perb[i];
  __syncthreads();
  for (int s = 16; s > 0; s >>= 1) {
    if (i < s) red[i] += red[i + s];
    __syncthreads();
  }
  if (i == 0) out[0] = red[0] * (1.0f / B_);
}

// ---------------------------------------------------------------------------
extern "C" void kernel_launch(void* const* d_in, const int* in_sizes, int n_in,
                              void* d_out, int out_size, void* d_ws, size_t ws_size,
                              hipStream_t stream) {
  const float* y   = (const float*)d_in[0];
  const float* x   = (const float*)d_in[1];
  const float* fw1 = (const float*)d_in[2];
  const float* fb1 = (const float*)d_in[3];
  const float* fw2 = (const float*)d_in[4];
  const float* fb2 = (const float*)d_in[5];
  const float* fw3 = (const float*)d_in[6];
  const float* fb3 = (const float*)d_in[7];
  const float* wih = (const float*)d_in[8];
  const float* whh = (const float*)d_in[9];
  const float* bih = (const float*)d_in[10];
  const float* bhh = (const float*)d_in[11];
  const float* hw  = (const float*)d_in[12];
  const float* hb  = (const float*)d_in[13];
  float* out = (float*)d_out;

  float* ws     = (float*)d_ws;
  float* z      = ws;                              // 16384*256
  float* logdet = z + (size_t)TOK_ * N_;           // 16384
  float* h1     = logdet + TOK_;                   // 16384*64
  float* h2     = h1 + (size_t)TOK_ * H_;          // 16384*64
  float* xg     = h2 + (size_t)TOK_ * H_;          // 16384*256 (reused twice)
  float* params = xg + (size_t)TOK_ * G4H_;        // 16384*1024
  float* perb   = params + (size_t)TOK_ * NP_;     // 32 (+pad to 32B align)

  // packed f16 weights (fragment-major); offsets in halves, 32B aligned
  _Float16* pwbase = (_Float16*)(perb + 64);
  _Float16* pw1    = pwbase;                       // 4*64*128   = 32768
  _Float16* pw2    = pw1 + 4 * H_ * HALF_;         // 4*64*64    = 16384
  _Float16* pw3    = pw2 + 4 * H_ * H_;            // 4*256*64   = 65536
  _Float16* pwih   = pw3 + 4 * N_ * H_;            // 2*256*64   = 32768
  _Float16* pwhh   = pwih + 2 * G4H_ * K_;         // 2*256*64   = 32768
  _Float16* pwhead = pwhh + 2 * G4H_ * H_;         // 1024*64    = 65536

  // one-time weight packing (treat layer dim as extra rows; tiles never
  // cross layer boundaries since per-layer rows are multiples of 16)
  pack_w_kernel<<<64, 256, 0, stream>>>(fw1, 4 * H_, HALF_, pw1);
  pack_w_kernel<<<32, 256, 0, stream>>>(fw2, 4 * H_, H_, pw2);
  pack_w_kernel<<<128, 256, 0, stream>>>(fw3, 4 * N_, H_, pw3);
  pack_w_kernel<<<64, 256, 0, stream>>>(wih, 2 * G4H_, K_, pwih);
  pack_w_kernel<<<64, 256, 0, stream>>>(whh, 2 * G4H_, H_, pwhh);
  pack_w_kernel<<<128, 256, 0, stream>>>(hw, NP_, H_, pwhead);

  flow_inverse_kernel<<<TOK_ / 32, 64, 0, stream>>>(y, pw1, pw2, pw3,
                                                    fb1, fb2, fb3, z, logdet);
  gates_kernel<<<TOK_ / 128, 256, 0, stream>>>(x, 1, pwih, bih, bhh, xg);
  lstm_rec_kernel<<<1, 256, 0, stream>>>(xg, pwhh, h1);
  gates_kernel<<<TOK_ / 128, 256, 0, stream>>>(h1, 0, pwih + G4H_ * K_,
                                               bih + G4H_, bhh + G4H_, xg);
  lstm_rec_kernel<<<1, 256, 0, stream>>>(xg, pwhh + G4H_ * H_, h2);
  head_kernel<<<TOK_ / 32, 256, 0, stream>>>(h2, pwhead, hb, params);
  kalman_kernel<<<B_, 256, 0, stream>>>(z, params, logdet, perb);
  final_reduce_kernel<<<1, 32, 0, stream>>>(perb, out);
}
